// GNN_NOCAT_NOWEIGHT_52699248722104
// MI455X (gfx1250) — compile-verified, matrix operand order checked by reading, not run on hardware
//
#include <hip/hip_runtime.h>
#include <hip/hip_bf16.h>
#include <stdint.h>

typedef __bf16 bf16;
typedef __attribute__((ext_vector_type(16))) __bf16 v16bf;
typedef __attribute__((ext_vector_type(8)))  __bf16 v8bf;
typedef __attribute__((ext_vector_type(8)))  float  v8f;

#define NSLOPE 0.2f

// ---------- device helpers ----------
__device__ __forceinline__ unsigned fkey(float f) {
  unsigned u = __float_as_uint(f);
  return (u & 0x80000000u) ? ~u : (u | 0x80000000u);
}
__device__ __forceinline__ float funkey(unsigned k) {
  unsigned u = (k & 0x80000000u) ? (k & 0x7FFFFFFFu) : ~k;
  return __uint_as_float(u);
}
__device__ __forceinline__ void edge_ji(const int* __restrict__ s, const int* __restrict__ d,
                                        int E, long long e, int& j, int& i) {
  if (e < (long long)E) { j = s[e]; i = d[e]; }
  else { j = i = (int)(e - E); }
}

// ---------- elementwise kernels ----------
__global__ void k_f32_to_bf16(const float* __restrict__ in, bf16* __restrict__ out, long long n) {
  long long t = (long long)blockIdx.x * blockDim.x + threadIdx.x;
  if (t < n) out[t] = (bf16)in[t];
}
__global__ void k_fill_f32(float* p, float v, long long n) {
  long long t = (long long)blockIdx.x * blockDim.x + threadIdx.x;
  if (t < n) p[t] = v;
}
__global__ void k_fill_u32(unsigned* p, unsigned v, long long n) {
  long long t = (long long)blockIdx.x * blockDim.x + threadIdx.x;
  if (t < n) p[t] = v;
}
__global__ void k_scale(float* p, float s, long long n) {
  long long t = (long long)blockIdx.x * blockDim.x + threadIdx.x;
  if (t < n) p[t] *= s;
}
__global__ void k_reduce_add(const float* __restrict__ x, int n, float* out) {
  int t = blockIdx.x * blockDim.x + threadIdx.x;
  if (t < n) atomicAdd(out, x[t]);
}

// ---------- B-operand pack: fp32 W[K,N] -> bf16 fragments in per-lane order ----------
// Pack layout: Bp[(((tn*(K/32) + kb)*32 + lane)*16 + e]
//   lane = g*16+r : column n = tn*16 + r ; element e -> k = kb*32 + (e<8 ? g*8+e : 16+g*8+e-8)
__global__ void k_pack_b(const float* __restrict__ W, bf16* __restrict__ Bp, int K, int N) {
  long long t = (long long)blockIdx.x * blockDim.x + threadIdx.x;
  if (t >= (long long)K * N) return;
  int e = (int)(t & 15);
  int lane = (int)((t >> 4) & 31);
  long long r2 = t >> 9;                 // = tn*(K/32) + kb
  int nkb = K >> 5;
  int kb = (int)(r2 % nkb), tn = (int)(r2 / nkb);
  int g = lane >> 4, r = lane & 15;
  int k = kb * 32 + ((e < 8) ? (g * 8 + e) : (16 + g * 8 + (e - 8)));
  int n = tn * 16 + r;
  Bp[t] = (bf16)W[(size_t)k * N + n];
}

// ---------- WMMA GEMM: C[M,N] = A[M,K](bf16) @ B[K,N] (+bias)(+acc)(+relu) ----------
// One wave per 16x64 C strip: one A fragment feeds 4 WMMAs. A fragment = two aligned
// b128 loads per k-step (zero-row guard for M edge); B fragment = one aligned v16bf
// load from the packed buffer. v_wmma_f32_16x16x32_bf16, fp32 accumulate.
__global__ __launch_bounds__(32)
void k_wmma_gemm(const bf16* __restrict__ A, const bf16* __restrict__ Bp,
                 const float* __restrict__ bias, float* __restrict__ C,
                 int M, int N, int K, int flags /*1=acc,2=relu*/,
                 const bf16* __restrict__ zrow) {
  int ng = N >> 6;                        // groups of four 16-wide n-tiles
  int tile = blockIdx.x;
  int tm = tile / ng, tg = tile - tm * ng;
  int lane = threadIdx.x & 31;
  int g = lane >> 4, r = lane & 15;
  int row = tm * 16 + r;
  const bf16* Arow = (row < M) ? (A + (size_t)row * K) : zrow;
  int nkb = K >> 5;
  v8f z = {0.f, 0.f, 0.f, 0.f, 0.f, 0.f, 0.f, 0.f};
  v8f acc[4] = {z, z, z, z};
  for (int kb = 0; kb < nkb; ++kb) {
    const bf16* ap = Arow + kb * 32 + g * 8;
    v8bf alo = *(const v8bf*)ap;
    v8bf ahi = *(const v8bf*)(ap + 16);
    v16bf av;
#pragma unroll
    for (int e = 0; e < 8; ++e) { av[e] = alo[e]; av[8 + e] = ahi[e]; }
#pragma unroll
    for (int t4 = 0; t4 < 4; ++t4) {
      int tn = tg * 4 + t4;
      const bf16* bp = Bp + (((size_t)tn * nkb + kb) * 32 + lane) * 16;
      v16bf bv = *(const v16bf*)bp;
      acc[t4] = __builtin_amdgcn_wmma_f32_16x16x32_bf16(false, av, false, bv,
                                                        (short)0, acc[t4], false, false);
    }
  }
#pragma unroll
  for (int t4 = 0; t4 < 4; ++t4) {
    int n = (tg * 4 + t4) * 16 + r;
    float bb = bias ? bias[n] : 0.f;
#pragma unroll
    for (int rr = 0; rr < 8; ++rr) {
      int m = tm * 16 + g * 8 + rr;
      if (m < M) {
        size_t idx = (size_t)m * N + n;
        float v = acc[t4][rr] + bb;
        if (flags & 1) v += C[idx];
        if (flags & 2) v = fmaxf(v, 0.f);
        C[idx] = v;
      }
    }
  }
}

// ---------- GAT / GATv2 kernels ----------
__global__ void k_att_dot(const float* __restrict__ hm, const float* __restrict__ att,
                          float* __restrict__ out, int Nn) {
  long long t = (long long)blockIdx.x * blockDim.x + threadIdx.x;
  if (t >= (long long)Nn * 3) return;
  int n = (int)(t / 3), h = (int)(t % 3);
  const float* hp = hm + (size_t)n * 192 + h * 64;
  const float* ap = att + h * 64;
  float s = 0.f;
#pragma unroll
  for (int c = 0; c < 64; ++c) s += hp[c] * ap[c];
  out[t] = s;
}

__global__ void k_gat_edge(const float* __restrict__ as_, const float* __restrict__ ad_,
                           const int* __restrict__ src, const int* __restrict__ dst,
                           int E, long long Etot, float* __restrict__ ebuf,
                           unsigned* __restrict__ mkey) {
  long long e = (long long)blockIdx.x * blockDim.x + threadIdx.x;
  if (e >= Etot) return;
  int j, i; edge_ji(src, dst, E, e, j, i);
#pragma unroll
  for (int h = 0; h < 3; ++h) {
    float v = as_[(size_t)j * 3 + h] + ad_[(size_t)i * 3 + h];
    v = v > 0.f ? v : NSLOPE * v;
    ebuf[e * 3 + h] = v;
    atomicMax(&mkey[(size_t)i * 3 + h], fkey(v));
  }
}

__global__ void k_gatv2_edge(const float* __restrict__ hl, const float* __restrict__ hr,
                             const float* __restrict__ att, const float* __restrict__ We,
                             const float* __restrict__ ea, const float* __restrict__ easum,
                             float invE, const int* __restrict__ src, const int* __restrict__ dst,
                             int E, long long Etot, float* __restrict__ ebuf,
                             unsigned* __restrict__ mkey) {
  long long t = (long long)blockIdx.x * blockDim.x + threadIdx.x;
  if (t >= Etot * 3) return;
  long long e = t / 3; int h = (int)(t % 3);
  int j, i; edge_ji(src, dst, E, e, j, i);
  bool he = (We != nullptr);
  float eav = 0.f;
  if (he) eav = (e < (long long)E) ? ea[e] : easum[0] * invE;
  const float* pl = hl + (size_t)j * 192 + h * 64;
  const float* pr = hr + (size_t)i * 192 + h * 64;
  const float* pa = att + h * 64;
  const float* pw = he ? (We + h * 64) : nullptr;
  float s = 0.f;
  for (int c = 0; c < 64; ++c) {
    float zz = pl[c] + pr[c];
    if (he) zz += eav * pw[c];
    zz = zz > 0.f ? zz : NSLOPE * zz;
    s += zz * pa[c];
  }
  ebuf[t] = s;
  atomicMax(&mkey[(size_t)i * 3 + h], fkey(s));
}

__global__ void k_edge_exp(const int* __restrict__ src, const int* __restrict__ dst,
                           int E, long long Etot, float* __restrict__ ebuf,
                           const unsigned* __restrict__ mkey, float* __restrict__ ssum) {
  long long t = (long long)blockIdx.x * blockDim.x + threadIdx.x;
  if (t >= Etot * 3) return;
  long long e = t / 3; int h = (int)(t % 3);
  int j, i; edge_ji(src, dst, E, e, j, i); (void)j;
  float m = funkey(mkey[(size_t)i * 3 + h]);
  if (!(m > -3.0e38f && m < 3.0e38f)) m = 0.f;  // non-finite -> 0 (matches ref)
  float ex = expf(ebuf[t] - m);
  ebuf[t] = ex;
  atomicAdd(&ssum[(size_t)i * 3 + h], ex);
}

__global__ void k_edge_scatter(const float* __restrict__ hs, const float* __restrict__ ebuf,
                               const float* __restrict__ ssum, const int* __restrict__ src,
                               const int* __restrict__ dst, int E, long long Etot,
                               float* __restrict__ outb) {
  long long t = (long long)blockIdx.x * blockDim.x + threadIdx.x;
  if (t >= Etot * 192) return;
  long long e = t / 192; int c = (int)(t - e * 192); int h = c >> 6;
  int j, i; edge_ji(src, dst, E, e, j, i);
  float alpha = ebuf[e * 3 + h] / (ssum[(size_t)i * 3 + h] + 1e-16f);
  atomicAdd(&outb[(size_t)i * 192 + c], hs[(size_t)j * 192 + c] * alpha);
}

__global__ void k_head_mean_add(const float* __restrict__ outb, const float* __restrict__ bias,
                                float* __restrict__ acc, int Nd) {
  long long t = (long long)blockIdx.x * blockDim.x + threadIdx.x;
  if (t >= (long long)Nd * 64) return;
  int n = (int)(t >> 6), c = (int)(t & 63);
  const float* p = outb + (size_t)n * 192;
  acc[t] += (p[c] + p[64 + c] + p[128 + c]) * (1.f / 3.f) + bias[c];
}

// ---------- SAGE kernels ----------
__global__ void k_sage_scatter(const float* __restrict__ xp, const int* __restrict__ src,
                               const int* __restrict__ dst, int E, int ds,
                               float* __restrict__ sbuf) {
  long long t = (long long)blockIdx.x * blockDim.x + threadIdx.x;
  if (t >= (long long)E * ds) return;
  long long e = t / ds; int c = (int)(t - e * ds);
  int j = src[e], i = dst[e];
  atomicAdd(&sbuf[(size_t)i * ds + c], xp[(size_t)j * ds + c]);
}
__global__ void k_sage_count(const int* __restrict__ dst, int E, float* __restrict__ cnt) {
  int t = blockIdx.x * blockDim.x + threadIdx.x;
  if (t < E) atomicAdd(&cnt[dst[t]], 1.f);
}
__global__ void k_sage_div(float* __restrict__ sbuf, const float* __restrict__ cnt,
                           int Nd, int ds) {
  long long t = (long long)blockIdx.x * blockDim.x + threadIdx.x;
  if (t >= (long long)Nd * ds) return;
  int n = (int)(t / ds);
  sbuf[t] /= fmaxf(cnt[n], 1.f);
}
__global__ void k_l2norm_add(const float* __restrict__ x, float* __restrict__ acc, int Nd) {
  int n = blockIdx.x * blockDim.x + threadIdx.x;
  if (n >= Nd) return;
  const float* p = x + (size_t)n * 64;
  float s = 0.f;
#pragma unroll
  for (int c = 0; c < 64; ++c) s += p[c] * p[c];
  float inv = 1.f / fmaxf(sqrtf(s), 1e-12f);
  float* a = acc + (size_t)n * 64;
#pragma unroll
  for (int c = 0; c < 64; ++c) a[c] += p[c] * inv;
}
__global__ void k_l2norm_out(const float* __restrict__ x, float* __restrict__ out, int Nd) {
  int n = blockIdx.x * blockDim.x + threadIdx.x;
  if (n >= Nd) return;
  const float* p = x + (size_t)n * 64;
  float s = 0.f;
#pragma unroll
  for (int c = 0; c < 64; ++c) s += p[c] * p[c];
  float inv = 1.f / fmaxf(sqrtf(s), 1e-12f);
  float* o = out + (size_t)n * 64;
#pragma unroll
  for (int c = 0; c < 64; ++c) o[c] = p[c] * inv;
}

// ---------- host orchestration ----------
static inline unsigned gsz(long long n, int b) { return (unsigned)((n + b - 1) / b); }

static void run_gemm(hipStream_t st, const bf16* A, const float* Wf, const float* bias,
                     float* C, int M, int N, int K, int flags, bf16* wbuf, const bf16* zrow) {
  long long wn = (long long)K * N;
  k_pack_b<<<gsz(wn, 256), 256, 0, st>>>(Wf, wbuf, K, N);
  long long tiles = (long long)((M + 15) / 16) * (N / 64);
  k_wmma_gemm<<<(unsigned)tiles, 32, 0, st>>>(A, wbuf, bias, C, M, N, K, flags, zrow);
}

enum { K_GAT = 0, K_GATV2 = 1, K_GATV2E = 2, K_SAGE = 3 };
struct RelDef { int kind, st, dt, E, ei; };
// Sorted by relation name (pytree dict-key order); ei = d_in index of edge list.
static const RelDef RELS[12] = {
  {K_GAT,    0, 0, 400000, 134},  // collab
  {K_GATV2,  0, 0, 400000, 137},  // follows
  {K_SAGE,   0, 2, 400000, 135},  // hta
  {K_SAGE,   1, 2, 600000, 138},  // htt
  {K_GATV2E, 0, 0, 400000, 136},  // lfm
  {K_GAT,    0, 0, 400000, 139},  // linked
  {K_GAT,    0, 0, 400000, 140},  // mus
  {K_GAT,    0, 0, 400000, 141},  // pers
  {K_SAGE,   2, 0, 400000, 142},  // ta
  {K_SAGE,   2, 1, 600000, 143},  // tt
  {K_SAGE,   1, 0, 800000, 144},  // wb
  {K_SAGE,   0, 1, 800000, 145},  // wi
};
static const int PCNT[4] = {5, 6, 7, 5};

extern "C" void kernel_launch(void* const* d_in, const int* in_sizes, int n_in,
                              void* d_out, int out_size, void* d_ws, size_t ws_size,
                              hipStream_t stream) {
  (void)in_sizes; (void)n_in; (void)out_size; (void)ws_size;
  const int NN[3] = {100000, 150000, 1000};       // artist, track, tag
  const float ACCN[3] = {8.f, 2.f, 2.f};          // relations targeting each type
  const int ch1[3] = {128, 128, 64}, ch2[3] = {64, 64, 64};

  auto F = [&](int i) -> const float* { return (const float*)d_in[i]; };

  // params pytree flat index bases: W1=4, W2=5, b1=6, b2=7, then layer1, layer2
  int pbase[2][12];
  { int idx = 8;
    for (int L = 0; L < 2; ++L)
      for (int r = 0; r < 12; ++r) { pbase[L][r] = idx; idx += PCNT[RELS[r].kind]; } }

  // ---- workspace bump allocator ----
  size_t off = 0;
  auto alloc = [&](size_t elems, size_t esize) -> void* {
    void* p = (char*)d_ws + off;
    off += (elems * esize + 255) & ~(size_t)255;
    return p;
  };
  bf16* xbf[3];  xbf[0]  = (bf16*)alloc((size_t)NN[0] * 128, 2);
                 xbf[1]  = (bf16*)alloc((size_t)NN[1] * 128, 2);
                 xbf[2]  = (bf16*)alloc((size_t)NN[2] * 64, 2);
  bf16* x1bf[3]; for (int t = 0; t < 3; ++t) x1bf[t] = (bf16*)alloc((size_t)NN[t] * 64, 2);
  float* acc1[3]; for (int t = 0; t < 3; ++t) acc1[t] = (float*)alloc((size_t)NN[t] * 64, 4);
  float* acc2[3]; for (int t = 0; t < 3; ++t) acc2[t] = (float*)alloc((size_t)NN[t] * 64, 4);
  float* tmpS = (float*)alloc(25600000, 4);   // hs/hl/xp, MLP hidden (100k x 256)
  float* tmpD = (float*)alloc(19200000, 4);   // hd/hr, SAGE mean buffer
  float* tmpO = (float*)alloc(19200000, 4);   // per-relation output buffer
  bf16*  abf  = (bf16*)alloc(25600000, 2);    // bf16 staging for GEMM A operands
  float* ebuf = (float*)alloc(1600000, 4);    // per-edge logits/exp (Etot*3)
  float* ssum = (float*)alloc(300000, 4);
  unsigned* mkey = (unsigned*)alloc(300000, 4);
  float* cnt  = (float*)alloc(150000, 4);
  float* avs  = (float*)alloc(450000, 4);
  float* avd  = (float*)alloc(300000, 4);
  bf16*  wbuf = (bf16*)alloc(65536, 2);       // packed bf16 weights (max 128x192)
  bf16*  zrow = (bf16*)alloc(256, 2);         // zero row for M-edge A loads
  float* scal = (float*)alloc(16, 4);

  k_fill_u32<<<1, 128, 0, stream>>>((unsigned*)zrow, 0u, 128);  // 256 bf16 zeros

  auto do_layer = [&](int L, bf16* const* xin, const int* ch, float* const* acc) {
    for (int t = 0; t < 3; ++t)
      k_fill_f32<<<gsz((long long)NN[t] * 64, 256), 256, 0, stream>>>(acc[t], 0.f, (long long)NN[t] * 64);
    for (int r = 0; r < 12; ++r) {
      const RelDef& R = RELS[r];
      int p = pbase[L][r];
      int ds = ch[R.st], dd = ch[R.dt];
      int Ns = NN[R.st], Nd = NN[R.dt];
      const int* ei  = (const int*)d_in[R.ei];
      const int* src = ei;
      const int* dst = ei + R.E;
      if (R.kind == K_SAGE) {  // leaves: Wl, Wp, Wr, bl, bp
        run_gemm(stream, xin[R.st], F(p + 1), F(p + 4), tmpS, Ns, ds, ds, 2, wbuf, zrow);  // xp=relu
        k_fill_f32<<<gsz((long long)Nd * ds, 256), 256, 0, stream>>>(tmpD, 0.f, (long long)Nd * ds);
        k_fill_f32<<<gsz(Nd, 256), 256, 0, stream>>>(cnt, 0.f, Nd);
        k_sage_scatter<<<gsz((long long)R.E * ds, 256), 256, 0, stream>>>(tmpS, src, dst, R.E, ds, tmpD);
        k_sage_count<<<gsz(R.E, 256), 256, 0, stream>>>(dst, R.E, cnt);
        k_sage_div<<<gsz((long long)Nd * ds, 256), 256, 0, stream>>>(tmpD, cnt, Nd, ds);
        k_f32_to_bf16<<<gsz((long long)Nd * ds, 256), 256, 0, stream>>>(tmpD, abf, (long long)Nd * ds);
        run_gemm(stream, abf,       F(p + 0), F(p + 3), tmpO, Nd, 64, ds, 0, wbuf, zrow);  // agg@Wl+bl
        run_gemm(stream, xin[R.dt], F(p + 2), nullptr,  tmpO, Nd, 64, dd, 1, wbuf, zrow);  // += xd@Wr
        k_l2norm_add<<<gsz(Nd, 128), 128, 0, stream>>>(tmpO, acc[R.dt], Nd);
      } else {
        long long Etot = (long long)R.E + Nd;  // + self loops
        const float* bias_;
        if (R.kind == K_GAT) {  // Wdst, Wsrc, att_dst, att_src, bias
          run_gemm(stream, xin[R.st], F(p + 1), nullptr, tmpS, Ns, 192, ds, 0, wbuf, zrow);  // hs
          run_gemm(stream, xin[R.dt], F(p + 0), nullptr, tmpD, Nd, 192, dd, 0, wbuf, zrow);  // hd
          k_att_dot<<<gsz((long long)Ns * 3, 256), 256, 0, stream>>>(tmpS, F(p + 3), avs, Ns);
          k_att_dot<<<gsz((long long)Nd * 3, 256), 256, 0, stream>>>(tmpD, F(p + 2), avd, Nd);
          bias_ = F(p + 4);
        } else {  // GATV2: Wl,Wr,att,bias,bl,br   GATV2E: We,Wl,Wr,att,bias,bl,br
          int b = (R.kind == K_GATV2E) ? p + 1 : p;
          run_gemm(stream, xin[R.st], F(b + 0), F(b + 3 + (R.kind == K_GATV2E)), tmpS, Ns, 192, ds, 0, wbuf, zrow); // hl+bl
          run_gemm(stream, xin[R.dt], F(b + 1), F(b + 4 + (R.kind == K_GATV2E)), tmpD, Nd, 192, dd, 0, wbuf, zrow); // hr+br
          bias_ = (R.kind == K_GATV2E) ? F(p + 4) : F(p + 3);
        }
        k_fill_u32<<<gsz((long long)Nd * 3, 256), 256, 0, stream>>>(mkey, 0u, (long long)Nd * 3);
        k_fill_f32<<<gsz((long long)Nd * 3, 256), 256, 0, stream>>>(ssum, 0.f, (long long)Nd * 3);
        k_fill_f32<<<gsz((long long)Nd * 192, 256), 256, 0, stream>>>(tmpO, 0.f, (long long)Nd * 192);
        if (R.kind == K_GAT) {
          k_gat_edge<<<gsz(Etot, 256), 256, 0, stream>>>(avs, avd, src, dst, R.E, Etot, ebuf, mkey);
        } else {
          const float* We = nullptr; float invE = 0.f;
          const float* att = (R.kind == K_GATV2E) ? F(p + 3) : F(p + 2);
          if (R.kind == K_GATV2E) {
            We = F(p); invE = 1.f / (float)R.E;
            k_fill_f32<<<1, 1, 0, stream>>>(scal, 0.f, 1);
            k_reduce_add<<<gsz(R.E, 256), 256, 0, stream>>>(F(3), R.E, scal);  // mean(edge_attr)
          }
          k_gatv2_edge<<<gsz(Etot * 3, 128), 128, 0, stream>>>(tmpS, tmpD, att, We, F(3), scal,
                                                               invE, src, dst, R.E, Etot, ebuf, mkey);
        }
        k_edge_exp<<<gsz(Etot * 3, 256), 256, 0, stream>>>(src, dst, R.E, Etot, ebuf, mkey, ssum);
        k_edge_scatter<<<gsz(Etot * 192, 256), 256, 0, stream>>>(tmpS, ebuf, ssum, src, dst, R.E, Etot, tmpO);
        k_head_mean_add<<<gsz((long long)Nd * 64, 256), 256, 0, stream>>>(tmpO, bias_, acc[R.dt], Nd);
      }
    }
    for (int t = 0; t < 3; ++t)
      k_scale<<<gsz((long long)NN[t] * 64, 256), 256, 0, stream>>>(acc[t], 1.f / ACCN[t], (long long)NN[t] * 64);
  };

  // ---- layer 1 ----
  k_f32_to_bf16<<<gsz((long long)NN[0] * 128, 256), 256, 0, stream>>>(F(0), xbf[0], (long long)NN[0] * 128);
  k_f32_to_bf16<<<gsz((long long)NN[1] * 128, 256), 256, 0, stream>>>(F(1), xbf[1], (long long)NN[1] * 128);
  k_f32_to_bf16<<<gsz((long long)NN[2] * 64, 256), 256, 0, stream>>>(F(2), xbf[2], (long long)NN[2] * 64);
  do_layer(0, xbf, ch1, acc1);

  // ---- layer 2 ----
  for (int t = 0; t < 3; ++t)
    k_f32_to_bf16<<<gsz((long long)NN[t] * 64, 256), 256, 0, stream>>>(acc1[t], x1bf[t], (long long)NN[t] * 64);
  do_layer(1, x1bf, ch2, acc2);

  // ---- MLP head on artist + final L2 normalize ----
  k_f32_to_bf16<<<gsz((long long)NN[0] * 64, 256), 256, 0, stream>>>(acc2[0], xbf[0], (long long)NN[0] * 64);
  run_gemm(stream, xbf[0], F(4), F(6), tmpS, NN[0], 256, 64, 2, wbuf, zrow);   // relu(x@W1+b1)
  k_f32_to_bf16<<<gsz((long long)NN[0] * 256, 256), 256, 0, stream>>>(tmpS, abf, (long long)NN[0] * 256);
  run_gemm(stream, abf, F(5), F(7), tmpO, NN[0], 64, 256, 0, wbuf, zrow);      // h@W2+b2
  k_l2norm_out<<<gsz(NN[0], 128), 128, 0, stream>>>(tmpO, (float*)d_out, NN[0]);
}
